// Encoder_72541997630027
// MI455X (gfx1250) — matrix operand order repaired and checked by reading, not compiled, not run on hardware
//
#include <hip/hip_runtime.h>
#include <hip/hip_bf16.h>
#include <math.h>

// ---------------------------------------------------------------------------
// CDNA5 (gfx1250, wave32) fused BiLSTM encoder.
//   Stage 1: per-block recurrence over 16 sequences, T=64, D=128, H=128.
//            gates(16x512) = [x_t | h](16x256) @ Wcomb^T via wmma bf16.
//   Stage 2: same engine for node-level BiLSTM (K=8, in=256, H2=64) with the
//            doc-embedding gather fused into the A-tile fill.
// ---------------------------------------------------------------------------

typedef __attribute__((ext_vector_type(16))) __bf16 v16bf;
typedef __attribute__((ext_vector_type(8)))  __bf16 v8bf;
typedef __attribute__((ext_vector_type(8)))  float  v8f;

union Frag { v16bf v; v8bf h[2]; };   // trivial union: 32B fragment = 2x16B LDS loads

__device__ __forceinline__ float sigmoidf_(float x) {
    return 1.0f / (1.0f + __expf(-x));
}

// ---- layer 1 geometry -------------------------------------------------------
#define L1_WSTR 264   // bf16 halfs per W row: 256 + 8 pad (row step = 132 banks -> conflict-free 16-row groups)
#define L1_ASTR 264   // A tile row stride ([x(128) | h(128)] + pad)
#define L1_GSTR 516   // f32 per gates row: 512 + 4 pad

__global__ __launch_bounds__(256) void lstm1_kernel(
    const float* __restrict__ docs,      // (4096,64,128)
    const int*   __restrict__ doc_lens,  // (4096)
    const float* __restrict__ Wih_f, const float* __restrict__ Whh_f,
    const float* __restrict__ Wih_b, const float* __restrict__ Whh_b,
    float* __restrict__ doc_emb)         // (4096,256) [fwd 0:128 | bwd 128:256]
{
    extern __shared__ __align__(16) char smem[];
    __bf16* Wl = (__bf16*)smem;                // 512 x L1_WSTR  bf16
    __bf16* Ab = Wl + 512 * L1_WSTR;           // 16  x L1_ASTR  bf16
    float*  Gt = (float*)(Ab + 16 * L1_ASTR);  // 16  x L1_GSTR  f32

    const int tid  = threadIdx.x;
    const int lane = tid & 31;
    const int wave = tid >> 5;
    const int dir  = blockIdx.y;
    const int base = blockIdx.x * 16;

    const float* Wih = dir ? Wih_b : Wih_f;
    const float* Whh = dir ? Whh_b : Whh_f;

    // Stage Wcomb = [Wih | Whh] (512 x 256) as bf16 into LDS (read once per block).
    for (int idx = tid; idx < 512 * 256; idx += 256) {
        int n = idx >> 8, k = idx & 255;
        float w = (k < 128) ? Wih[n * 128 + k] : Whh[n * 128 + (k - 128)];
        Wl[n * L1_WSTR + k] = (__bf16)w;
    }

    const int m   = tid & 15;              // batch row within the 16-row tile
    const int nb  = (tid >> 4) * 8;        // hidden-base for elementwise (0..120)
    const int len = doc_lens[base + m];

    float c[8], hsum[8];
    #pragma unroll
    for (int j = 0; j < 8; ++j) { c[j] = 0.f; hsum[j] = 0.f; }
    #pragma unroll
    for (int j = 0; j < 8; ++j) Ab[m * L1_ASTR + 128 + nb + j] = (__bf16)0.f; // h0 = 0

    const int fm = lane & 15;              // fragment row (A) / col (B)
    const int hi = lane >> 4;              // lane half select

    for (int t = 0; t < 64; ++t) {
        // ---- fill x region of A (cols 0..127) ------------------------------
        int tsrc = dir ? min(max(len - 1 - t, 0), 63) : t;
        const float* xp = docs + (((size_t)(base + m)) * 64 + tsrc) * 128 + nb;
        #pragma unroll
        for (int j = 0; j < 8; ++j)
            Ab[m * L1_ASTR + nb + j] = (__bf16)xp[j];
        if (t < 63) {                       // prefetch next step's row (L2-resident)
            int tn = dir ? min(max(len - 2 - t, 0), 63) : (t + 1);
            __builtin_prefetch(docs + (((size_t)(base + m)) * 64 + tn) * 128 + nb, 0, 3);
        }
        __syncthreads();

        // ---- gates(16x512) = A(16x256) @ Wcomb^T ; wave w -> N-tiles 4w..4w+3
        v8f acc0 = {}, acc1 = {}, acc2 = {}, acc3 = {};
        #pragma unroll
        for (int kt = 0; kt < 8; ++kt) {
            Frag a;   // A layout: element e -> K = kt*32 + (e>>3)*16 + hi*8 + (e&7)
            const __bf16* ap = Ab + fm * L1_ASTR + kt * 32 + hi * 8;
            a.h[0] = *(const v8bf*)(ap);
            a.h[1] = *(const v8bf*)(ap + 16);
            #pragma unroll
            for (int s = 0; s < 4; ++s) {
                int nt = wave * 4 + s;
                Frag b;  // B layout: lane col = fm, K = kt*32 + (e>>3)*16 + hi*8 + (e&7)
                const __bf16* bp = Wl + (nt * 16 + fm) * L1_WSTR + kt * 32 + hi * 8;
                b.h[0] = *(const v8bf*)(bp);
                b.h[1] = *(const v8bf*)(bp + 16);
                v8f& acc = (s == 0) ? acc0 : (s == 1) ? acc1 : (s == 2) ? acc2 : acc3;
                acc = __builtin_amdgcn_wmma_f32_16x16x32_bf16(
                        false, a.v, false, b.v, (short)0, acc, false, false);
            }
        }
        // ---- scatter D tiles (m = hi*8+v, n = nt*16+fm) into gates LDS -----
        #pragma unroll
        for (int s = 0; s < 4; ++s) {
            int nt = wave * 4 + s;
            const v8f& acc = (s == 0) ? acc0 : (s == 1) ? acc1 : (s == 2) ? acc2 : acc3;
            #pragma unroll
            for (int v = 0; v < 8; ++v)
                Gt[(hi * 8 + v) * L1_GSTR + nt * 16 + fm] = acc[v];
        }
        __syncthreads();

        // ---- elementwise LSTM cell; c/hsum live in registers ---------------
        #pragma unroll
        for (int j = 0; j < 8; ++j) {
            int n = nb + j;
            float gi = Gt[m * L1_GSTR + n];
            float gf = Gt[m * L1_GSTR + 128 + n];
            float gg = Gt[m * L1_GSTR + 256 + n];
            float go = Gt[m * L1_GSTR + 384 + n];
            float cn = sigmoidf_(gf) * c[j] + sigmoidf_(gi) * tanhf(gg);
            float hn = sigmoidf_(go) * tanhf(cn);
            c[j] = cn;
            if (t < len) hsum[j] += hn;   // masked mean; rev-scatter is a bijection on [0,len)
            Ab[m * L1_ASTR + 128 + n] = (__bf16)hn;
        }
        // no barrier needed: next x-fill touches a disjoint A region; the
        // pre-WMMA barrier orders both writes against the next GEMM.
    }

    float inv = 1.0f / (float)len;
    #pragma unroll
    for (int j = 0; j < 8; ++j)
        doc_emb[((size_t)(base + m)) * 256 + dir * 128 + nb + j] = hsum[j] * inv;
}

// ---- layer 2 geometry -------------------------------------------------------
#define L2_WSTR 328   // 320 + 8 pad
#define L2_ASTR 328
#define L2_GSTR 260   // 256 + 4 pad

__global__ __launch_bounds__(256) void lstm2_kernel(
    const float* __restrict__ doc_emb,     // (4096,256)
    const int*   __restrict__ node_doc_idx,// (512,8) flat
    const int*   __restrict__ node_lens,   // (512)   flat
    const int*   __restrict__ gnn,         // (8)
    const float* __restrict__ Wih_f, const float* __restrict__ Whh_f,
    const float* __restrict__ Wih_b, const float* __restrict__ Whh_b,
    float* __restrict__ out)               // (512,128) masked node_emb
{
    extern __shared__ __align__(16) char smem[];
    __bf16* Wl = (__bf16*)smem;                // 256 x L2_WSTR bf16
    __bf16* Ab = Wl + 256 * L2_WSTR;           // 16  x L2_ASTR bf16 ([x 256 | h 64])
    float*  Gt = (float*)(Ab + 16 * L2_ASTR);  // 16  x L2_GSTR f32

    const int tid  = threadIdx.x;
    const int lane = tid & 31;
    const int wave = tid >> 5;
    const int dir  = blockIdx.y;
    const int base = blockIdx.x * 16;

    const float* Wih = dir ? Wih_b : Wih_f;
    const float* Whh = dir ? Whh_b : Whh_f;

    // Wcomb2 = [Wih2 (256x256) | Whh2 (256x64)] -> 256 x 320 bf16
    for (int idx = tid; idx < 256 * 320; idx += 256) {
        int n = idx / 320, k = idx - n * 320;
        float w = (k < 256) ? Wih[n * 256 + k] : Whh[n * 64 + (k - 256)];
        Wl[n * L2_WSTR + k] = (__bf16)w;
    }

    const int m    = tid & 15;             // node row within tile
    const int row  = base + m;             // flat node row in [0,512)
    const int nb   = (tid >> 4) * 4;       // hidden base (0..60), 4 per thread
    const int cb   = (tid >> 4) * 16;      // x-fill col base (0..240), 16 per thread
    const int nlen = node_lens[row];

    float c[4], hsum[4];
    #pragma unroll
    for (int j = 0; j < 4; ++j) { c[j] = 0.f; hsum[j] = 0.f; }
    #pragma unroll
    for (int j = 0; j < 4; ++j) Ab[m * L2_ASTR + 256 + nb + j] = (__bf16)0.f;

    const int fm = lane & 15;
    const int hi = lane >> 4;

    for (int t = 0; t < 8; ++t) {
        // ---- fused gather: x_t = doc_emb[node_doc_idx[row, tsrc]] ----------
        int tsrc = dir ? min(max(nlen - 1 - t, 0), 7) : t;
        int didx = node_doc_idx[row * 8 + tsrc];
        const float* xp = doc_emb + ((size_t)didx) * 256 + cb;
        #pragma unroll
        for (int j = 0; j < 16; ++j)
            Ab[m * L2_ASTR + cb + j] = (__bf16)xp[j];
        __syncthreads();

        // ---- gates(16x256) = A(16x320) @ Wcomb^T ; wave w -> N-tiles 2w..2w+1
        v8f acc0 = {}, acc1 = {};
        #pragma unroll
        for (int kt = 0; kt < 10; ++kt) {
            Frag a;
            const __bf16* ap = Ab + fm * L2_ASTR + kt * 32 + hi * 8;
            a.h[0] = *(const v8bf*)(ap);
            a.h[1] = *(const v8bf*)(ap + 16);
            #pragma unroll
            for (int s = 0; s < 2; ++s) {
                int nt = wave * 2 + s;
                Frag b;
                const __bf16* bp = Wl + (nt * 16 + fm) * L2_WSTR + kt * 32 + hi * 8;
                b.h[0] = *(const v8bf*)(bp);
                b.h[1] = *(const v8bf*)(bp + 16);
                v8f& acc = (s == 0) ? acc0 : acc1;
                acc = __builtin_amdgcn_wmma_f32_16x16x32_bf16(
                        false, a.v, false, b.v, (short)0, acc, false, false);
            }
        }
        #pragma unroll
        for (int s = 0; s < 2; ++s) {
            int nt = wave * 2 + s;
            const v8f& acc = (s == 0) ? acc0 : acc1;
            #pragma unroll
            for (int v = 0; v < 8; ++v)
                Gt[(hi * 8 + v) * L2_GSTR + nt * 16 + fm] = acc[v];
        }
        __syncthreads();

        #pragma unroll
        for (int j = 0; j < 4; ++j) {
            int n = nb + j;
            float gi = Gt[m * L2_GSTR + n];
            float gf = Gt[m * L2_GSTR + 64 + n];
            float gg = Gt[m * L2_GSTR + 128 + n];
            float go = Gt[m * L2_GSTR + 192 + n];
            float cn = sigmoidf_(gf) * c[j] + sigmoidf_(gi) * tanhf(gg);
            float hn = sigmoidf_(go) * tanhf(cn);
            c[j] = cn;
            if (t < nlen) hsum[j] += hn;
            Ab[m * L2_ASTR + 256 + n] = (__bf16)hn;
        }
    }

    // masked mean -> output (gmask applied here; rows past num_nodes write 0)
    int   g    = row >> 6, nn = row & 63;
    float msk  = (nn < gnn[g]) ? 1.0f : 0.0f;
    float inv  = msk / (float)nlen;
    #pragma unroll
    for (int j = 0; j < 4; ++j)
        out[(size_t)row * 128 + dir * 64 + nb + j] = hsum[j] * inv;
}

__global__ void gmask_kernel(const int* __restrict__ gnn, float* __restrict__ out_mask) {
    int idx = blockIdx.x * 256 + threadIdx.x;   // 512 (G*N) entries
    if (idx < 512)
        out_mask[idx] = ((idx & 63) < gnn[idx >> 6]) ? 1.0f : 0.0f;
}

extern "C" void kernel_launch(void* const* d_in, const int* in_sizes, int n_in,
                              void* d_out, int out_size, void* d_ws, size_t ws_size,
                              hipStream_t stream) {
    const float* docs         = (const float*)d_in[0];
    const int*   doc_lens     = (const int*)  d_in[1];
    const int*   node_doc_idx = (const int*)  d_in[2];
    const int*   node_lens    = (const int*)  d_in[3];
    const int*   gnn          = (const int*)  d_in[4];
    const float* Wih1f = (const float*)d_in[5];
    const float* Whh1f = (const float*)d_in[6];
    const float* Wih1b = (const float*)d_in[7];
    const float* Whh1b = (const float*)d_in[8];
    const float* Wih2f = (const float*)d_in[9];
    const float* Whh2f = (const float*)d_in[10];
    const float* Wih2b = (const float*)d_in[11];
    const float* Whh2b = (const float*)d_in[12];

    float* out     = (float*)d_out;          // 512*128 node_emb + 512 gmask
    float* doc_emb = (float*)d_ws;           // 4096*256 f32 scratch

    size_t lds1 = (size_t)(512 * L1_WSTR + 16 * L1_ASTR) * sizeof(__bf16)
                + (size_t)16 * L1_GSTR * sizeof(float);   // ~305 KB of 320 KB WGP LDS
    size_t lds2 = (size_t)(256 * L2_WSTR + 16 * L2_ASTR) * sizeof(__bf16)
                + (size_t)16 * L2_GSTR * sizeof(float);   // ~191 KB

    lstm1_kernel<<<dim3(256, 2), 256, lds1, stream>>>(
        docs, doc_lens, Wih1f, Whh1f, Wih1b, Whh1b, doc_emb);
    lstm2_kernel<<<dim3(32, 2), 256, lds2, stream>>>(
        doc_emb, node_doc_idx, node_lens, gnn, Wih2f, Whh2f, Wih2b, Whh2b, out);
    gmask_kernel<<<2, 256, 0, stream>>>(gnn, out + 512 * 128);
}
